// GATBlock_50620484551056
// MI455X (gfx1250) — compile-verified, matrix operand order checked by reading, not run on hardware
//
#include <hip/hip_runtime.h>
#include <stdint.h>

#define D_DIM 1024
#define H_HEADS 8
#define C_CH 128
#define NEG_SLOPE 0.2f
#define BN_EPS 1e-5f

typedef __bf16        v16bf __attribute__((ext_vector_type(16)));
typedef float         v8f   __attribute__((ext_vector_type(8)));
typedef unsigned int  v4u   __attribute__((ext_vector_type(4)));
typedef unsigned int  u32x4 __attribute__((ext_vector_type(4)));
typedef int           i32x8v __attribute__((ext_vector_type(8)));
typedef int           i32x4v __attribute__((ext_vector_type(4)));

union Frag {
    struct { v4u a, b; } q;   // two 16B chunks = 16 bf16
    v16bf v;
};

__device__ __forceinline__ unsigned short f2bf(float f) {
    unsigned int u = __float_as_uint(f);
    u += 0x7FFFu + ((u >> 16) & 1u);     // round-to-nearest-even
    return (unsigned short)(u >> 16);
}
__device__ __forceinline__ float bf2f(unsigned short h) {
    return __uint_as_float(((unsigned int)h) << 16);
}
__device__ __forceinline__ float leaky(float e) { return e >= 0.f ? e : NEG_SLOPE * e; }
// monotone float <-> uint mapping for atomicMax on floats
__device__ __forceinline__ unsigned fkey(float x) {
    unsigned u = __float_as_uint(x);
    return (u & 0x80000000u) ? ~u : (u | 0x80000000u);
}
__device__ __forceinline__ float funkey(unsigned k) {
    return __uint_as_float((k & 0x80000000u) ? (k ^ 0x80000000u) : ~k);
}

// ---------------- conversion: f32 -> bf16 hi/lo split ----------------
__global__ void k_split_bf16(const float* __restrict__ src,
                             unsigned short* __restrict__ hi,
                             unsigned short* __restrict__ lo, int n) {
    int i = blockIdx.x * blockDim.x + threadIdx.x;
    int stride = gridDim.x * blockDim.x;
    for (; i < n; i += stride) {
        float f = src[i];
        unsigned short h = f2bf(f);
        hi[i] = h;
        lo[i] = f2bf(f - bf2f(h));
    }
}

// ---------------- W transpose + bf16 hi/lo split (LDS tiled) ----------------
__global__ __launch_bounds__(256) void k_wt_split(const float* __restrict__ W,
                                                  unsigned short* __restrict__ hi,
                                                  unsigned short* __restrict__ lo) {
    __shared__ float tile[32][33];
    const int k0 = blockIdx.y * 32;     // W row block
    const int n0 = blockIdx.x * 32;     // W col block
    const int tx = threadIdx.x;         // 0..31
    const int ty = threadIdx.y;         // 0..7
    for (int r = ty; r < 32; r += 8)
        tile[r][tx] = W[(size_t)(k0 + r) * D_DIM + n0 + tx];
    __syncthreads();
    for (int r = ty; r < 32; r += 8) {
        float f = tile[tx][r];          // = W[k0+tx][n0+r]
        unsigned short h = f2bf(f);
        size_t o = (size_t)(n0 + r) * D_DIM + k0 + tx;   // Wt[n][k]
        hi[o] = h;
        lo[o] = f2bf(f - bf2f(h));
    }
}

// ---------------- GEMM: xl = x @ W via bf16 WMMA, split-precision ----------------
// Block = 256 threads = 8 waves; block tile 128x128; wave w owns rows [16w,16w+16).
// B tile (Wt, 128 cols x 32 K, hi+lo) is staged into LDS once per K-step by the
// Tensor Data Mover (tensor_load_to_lds, wave 0), with 16B LDS padding per 64B row
// so ds_load_b128 fragment reads hit distinct banks. A frags come straight from
// global (no cross-wave reuse). 3 WMMAs per 16x16 tile give ~fp32 accuracy.
#define BROW 40   // 32 bf16 + 8 pad => 80B row stride in LDS

__global__ __launch_bounds__(256) void k_gemm_wmma(
    const unsigned short* __restrict__ xh, const unsigned short* __restrict__ xlo,
    const unsigned short* __restrict__ wh, const unsigned short* __restrict__ wlo,
    float* __restrict__ out, int nrows) {
    __shared__ __align__(16) unsigned short sBh[128 * BROW];
    __shared__ __align__(16) unsigned short sBl[128 * BROW];

    const int lane = threadIdx.x & 31;
    const int wave = threadIdx.x >> 5;
    const int half = lane >> 4;    // K-half select
    const int r16  = lane & 15;

    const int m_base = blockIdx.x * 128 + wave * 16;
    const int n_base = blockIdx.y * 128;

    int arow = m_base + r16;
    if (arow >= nrows) arow = nrows - 1;   // clamp loads; stores guarded below

    v8f acc[8] = {};

    const unsigned short* pAh = xh  + (size_t)arow * D_DIM + half * 8;
    const unsigned short* pAl = xlo + (size_t)arow * D_DIM + half * 8;

    // ---- TDM descriptor (CDNA5 ISA ch.8): 2-D tile, 128 rows x 32 elem, 2B data ----
    const unsigned ldsBh = (unsigned)(uintptr_t)(void*)sBh;
    const unsigned ldsBl = (unsigned)(uintptr_t)(void*)sBl;
    unsigned long long gah = (unsigned long long)(uintptr_t)(wh  + (size_t)n_base * D_DIM);
    unsigned long long gal = (unsigned long long)(uintptr_t)(wlo + (size_t)n_base * D_DIM);

    i32x8v g1;
    g1[0] = (int)((1u << 16)      // data_size = 2 bytes
                | (1u << 20)      // pad_enable
                | (3u << 22)      // pad_interval: every 16 DWORDs (64B row)
                | (3u << 25));    // pad_amount: 4 DWORDs (16B)
    g1[1] = (int)(1024u << 16);   // tensor_dim0 = 1024 (K extent)
    g1[2] = (int)(1024u << 16);   // tensor_dim1 = 1024 (rows)
    g1[3] = (int)(32u << 16);     // tile_dim0 = 32
    g1[4] = 128;                  // tile_dim1 = 128, tile_dim2 = 0
    g1[5] = 1024;                 // tensor_dim0_stride = 1024
    g1[6] = 0;                    // (stride0 hi | stride1 lo16 = 0)
    g1[7] = 16;                   // tensor_dim1_stride = 1M >> 16
    const i32x4v gz4 = {0, 0, 0, 0};
    const i32x8v gz8 = {0, 0, 0, 0, 0, 0, 0, 0};

    for (int k0 = 0; k0 < D_DIM; k0 += 32) {
        __syncthreads();          // previous step's LDS reads complete
        if (wave == 0) {
            u32x4 g0;
            g0.x = 1u;            // count=1, user descriptor
            g0.y = ldsBh;
            g0.z = (unsigned)gah;
            g0.w = ((unsigned)((gah >> 32) & 0x01FFFFFFull)) | 0x80000000u;  // type=2
            __builtin_amdgcn_tensor_load_to_lds(g0, g1, gz4, gz4, gz8, 0);
            g0.y = ldsBl;
            g0.z = (unsigned)gal;
            g0.w = ((unsigned)((gal >> 32) & 0x01FFFFFFull)) | 0x80000000u;
            __builtin_amdgcn_tensor_load_to_lds(g0, g1, gz4, gz4, gz8, 0);
            __builtin_amdgcn_s_wait_tensorcnt(0);
        }
        __syncthreads();          // B tile visible to all waves

        Frag ah, al;
        ah.q.a = *(const v4u*)(pAh);
        ah.q.b = *(const v4u*)(pAh + 16);
        al.q.a = *(const v4u*)(pAl);
        al.q.b = *(const v4u*)(pAl + 16);
#pragma unroll
        for (int j = 0; j < 8; ++j) {
            const int boff = (j * 16 + r16) * BROW + half * 8;
            Frag bh, bl;
            bh.q.a = *(const v4u*)(sBh + boff);
            bh.q.b = *(const v4u*)(sBh + boff + 16);
            bl.q.a = *(const v4u*)(sBl + boff);
            bl.q.b = *(const v4u*)(sBl + boff + 16);
            acc[j] = __builtin_amdgcn_wmma_f32_16x16x32_bf16(false, ah.v, false, bh.v,
                                                             (short)0, acc[j], false, false);
            acc[j] = __builtin_amdgcn_wmma_f32_16x16x32_bf16(false, ah.v, false, bl.v,
                                                             (short)0, acc[j], false, false);
            acc[j] = __builtin_amdgcn_wmma_f32_16x16x32_bf16(false, al.v, false, bh.v,
                                                             (short)0, acc[j], false, false);
        }
        pAh += 32;
        pAl += 32;
        gah += 64;
        gal += 64;
    }

#pragma unroll
    for (int j = 0; j < 8; ++j) {
        const int col = n_base + j * 16 + r16;
#pragma unroll
        for (int rr = 0; rr < 8; ++rr) {
            const int row = m_base + rr + half * 8;   // C layout: VGPR rr -> M=rr (+8 hi half)
            if (row < nrows) out[(size_t)row * D_DIM + col] = acc[j][rr];
        }
    }
}

// ---------------- per-node attention dots: a_src/a_dst [N,H] ----------------
__global__ __launch_bounds__(256) void k_att(const float* __restrict__ xl,
        const float* __restrict__ att_src, const float* __restrict__ att_dst,
        float* __restrict__ asrc, float* __restrict__ adst) {
    const int n = blockIdx.x;
    const int h = threadIdx.x >> 5;       // 8 waves = 8 heads
    const int lane = threadIdx.x & 31;
    const float4* xr = (const float4*)(xl + (size_t)n * D_DIM + h * C_CH);
    const float4* ws = (const float4*)(att_src + h * C_CH);
    const float4* wd = (const float4*)(att_dst + h * C_CH);
    float4 v = xr[lane], s = ws[lane], d = wd[lane];
    float ps = v.x * s.x + v.y * s.y + v.z * s.z + v.w * s.w;
    float pd = v.x * d.x + v.y * d.y + v.z * d.z + v.w * d.w;
#pragma unroll
    for (int off = 16; off > 0; off >>= 1) {
        ps += __shfl_down(ps, off, 32);
        pd += __shfl_down(pd, off, 32);
    }
    if (lane == 0) { asrc[n * H_HEADS + h] = ps; adst[n * H_HEADS + h] = pd; }
}

__device__ __forceinline__ void edge_sd(const int* ei, int e, int E, int& s, int& d) {
    if (e < E) { s = ei[e]; d = ei[E + e]; } else { s = d = e - E; }   // self-loops appended
}

// ---------------- edge pass 1: segment max (atomicMax on ordered keys) ----------------
__global__ void k_edge_max(const int* __restrict__ ei, const float* __restrict__ asrc,
                           const float* __restrict__ adst, unsigned* __restrict__ mkey,
                           int E, int nE) {
    int t = blockIdx.x * blockDim.x + threadIdx.x;
    if (t >= nE * H_HEADS) return;
    int e = t >> 3, h = t & 7, s, d;
    edge_sd(ei, e, E, s, d);
    float ev = leaky(asrc[s * H_HEADS + h] + adst[d * H_HEADS + h]);
    atomicMax(&mkey[d * H_HEADS + h], fkey(ev));
}

// ---------------- edge pass 2: segment sum of exp ----------------
__global__ void k_edge_sum(const int* __restrict__ ei, const float* __restrict__ asrc,
                           const float* __restrict__ adst, const unsigned* __restrict__ mkey,
                           float* __restrict__ denom, int E, int nE) {
    int t = blockIdx.x * blockDim.x + threadIdx.x;
    if (t >= nE * H_HEADS) return;
    int e = t >> 3, h = t & 7, s, d;
    edge_sd(ei, e, E, s, d);
    float ev = leaky(asrc[s * H_HEADS + h] + adst[d * H_HEADS + h]);
    float m = funkey(mkey[d * H_HEADS + h]);
    atomicAdd(&denom[d * H_HEADS + h], __expf(ev - m));
}

// ---------------- edge pass 3: alpha-weighted scatter-add (block per edge) ----------------
__global__ __launch_bounds__(256) void k_edge_aggr(const int* __restrict__ ei,
        const float* __restrict__ asrc, const float* __restrict__ adst,
        const unsigned* __restrict__ mkey, const float* __restrict__ denom,
        const float* __restrict__ xl, float* __restrict__ outag, int E, int nE) {
    __shared__ float s_alpha[H_HEADS];
    const int e = blockIdx.x;
    int s, d;
    edge_sd(ei, e, E, s, d);
    if (threadIdx.x < H_HEADS) {
        int h = threadIdx.x;
        float ev = leaky(asrc[s * H_HEADS + h] + adst[d * H_HEADS + h]);
        float m = funkey(mkey[d * H_HEADS + h]);
        s_alpha[h] = __expf(ev - m) / (denom[d * H_HEADS + h] + 1e-16f);
    }
    __syncthreads();
    const float* xr = xl + (size_t)s * D_DIM;
    float* orow = outag + (size_t)d * D_DIM;
#pragma unroll
    for (int i = 0; i < 4; ++i) {
        int c = threadIdx.x + 256 * i;
        atomicAdd(&orow[c], s_alpha[c >> 7] * xr[c]);
    }
}

// ---------------- h = prev + agg + bias (in-place), per-column sum/sumsq ----------------
__global__ __launch_bounds__(256) void k_bnstats(const float* __restrict__ prev,
        float* __restrict__ hbuf, const float* __restrict__ bias,
        float* __restrict__ sum, float* __restrict__ sumsq, int nrows) {
    const int col = blockIdx.x * 256 + threadIdx.x;
    const int r0 = blockIdx.y * 64;
    const float b = bias[col];
    float s = 0.f, s2 = 0.f;
    for (int r = 0; r < 64; ++r) {
        int row = r0 + r;
        if (row < nrows) {
            size_t idx = (size_t)row * D_DIM + col;
            float h = prev[idx] + hbuf[idx] + b;
            hbuf[idx] = h;
            s += h; s2 += h * h;
        }
    }
    atomicAdd(&sum[col], s);
    atomicAdd(&sumsq[col], s2);
}

// ---------------- BN normalize + ReLU ----------------
__global__ void k_bnapply(const float* __restrict__ hbuf, const float* __restrict__ sum,
                          const float* __restrict__ sumsq, const float* __restrict__ gamma,
                          const float* __restrict__ beta, float* __restrict__ out,
                          int total, float invN) {
    int i = blockIdx.x * blockDim.x + threadIdx.x;
    int stride = gridDim.x * blockDim.x;
    for (; i < total; i += stride) {
        int c = i & (D_DIM - 1);
        float mu = sum[c] * invN;
        float var = sumsq[c] * invN - mu * mu;
        float hn = (hbuf[i] - mu) * rsqrtf(var + BN_EPS) * gamma[c] + beta[c];
        out[i] = hn > 0.f ? hn : 0.f;
    }
}

extern "C" void kernel_launch(void* const* d_in, const int* in_sizes, int n_in,
                              void* d_out, int out_size, void* d_ws, size_t ws_size,
                              hipStream_t stream) {
    const float* prev    = (const float*)d_in[0];
    const float* x       = (const float*)d_in[1];
    const int*   ei      = (const int*)d_in[2];
    const float* W       = (const float*)d_in[3];
    const float* att_src = (const float*)d_in[4];
    const float* att_dst = (const float*)d_in[5];
    const float* bias    = (const float*)d_in[6];
    const float* gamma   = (const float*)d_in[7];
    const float* beta    = (const float*)d_in[8];

    const int N  = in_sizes[0] / D_DIM;
    const int E  = in_sizes[2] / 2;
    const int nE = E + N;
    const size_t ND = (size_t)N * D_DIM;

    char* p = (char*)d_ws;
    auto carve = [&](size_t bytes) -> void* {
        void* r = (void*)p;
        p += (bytes + 255) & ~(size_t)255;
        return r;
    };
    unsigned short* xh  = (unsigned short*)carve(ND * 2);
    unsigned short* xlo = (unsigned short*)carve(ND * 2);
    unsigned short* wth = (unsigned short*)carve((size_t)D_DIM * D_DIM * 2);
    unsigned short* wtl = (unsigned short*)carve((size_t)D_DIM * D_DIM * 2);
    float*    xl    = (float*)carve(ND * 4);
    float*    hbuf  = (float*)carve(ND * 4);
    float*    asrc  = (float*)carve((size_t)N * H_HEADS * 4);
    float*    adst  = (float*)carve((size_t)N * H_HEADS * 4);
    unsigned* mkey  = (unsigned*)carve((size_t)N * H_HEADS * 4);
    float*    denom = (float*)carve((size_t)N * H_HEADS * 4);
    float*    sum   = (float*)carve(D_DIM * 4);
    float*    sumsq = (float*)carve(D_DIM * 4);

    (void)hipMemsetAsync(hbuf, 0, ND * 4, stream);
    (void)hipMemsetAsync(mkey, 0, (size_t)N * H_HEADS * 4, stream);
    (void)hipMemsetAsync(denom, 0, (size_t)N * H_HEADS * 4, stream);
    (void)hipMemsetAsync(sum, 0, D_DIM * 4, stream);
    (void)hipMemsetAsync(sumsq, 0, D_DIM * 4, stream);

    k_split_bf16<<<512, 256, 0, stream>>>(x, xh, xlo, (int)ND);
    k_wt_split<<<dim3(D_DIM / 32, D_DIM / 32), dim3(32, 8), 0, stream>>>(W, wth, wtl);

    k_gemm_wmma<<<dim3((N + 127) / 128, D_DIM / 128), 256, 0, stream>>>(xh, xlo, wth, wtl, xl, N);

    k_att<<<N, 256, 0, stream>>>(xl, att_src, att_dst, asrc, adst);

    const int tE = nE * H_HEADS;
    k_edge_max<<<(tE + 255) / 256, 256, 0, stream>>>(ei, asrc, adst, mkey, E, nE);
    k_edge_sum<<<(tE + 255) / 256, 256, 0, stream>>>(ei, asrc, adst, mkey, denom, E, nE);
    k_edge_aggr<<<nE, 256, 0, stream>>>(ei, asrc, adst, mkey, denom, xl, hbuf, E, nE);

    k_bnstats<<<dim3(D_DIM / 256, (N + 63) / 64), 256, 0, stream>>>(prev, hbuf, bias, sum, sumsq, N);
    k_bnapply<<<2048, 256, 0, stream>>>(hbuf, sum, sumsq, gamma, beta, (float*)d_out,
                                        (int)ND, 1.0f / (float)N);
}
